// DiscreteQKTRBlock_25520695673113
// MI455X (gfx1250) — matrix operand order repaired and checked by reading, not compiled, not run on hardware
//
#include <hip/hip_runtime.h>
#include <stdint.h>

// ---------------------------------------------------------------------------
// DiscreteQKTRBlock for MI455X (gfx1250, wave32, WMMA bf16, async-to-LDS, TDM)
//
// N=100000 pts, PLANES=128, VEC=16, K=27 offsets.
// scores are rank-1: dq[n,c] = codebook[c]*choice[n]  =>
//   score[k,n] = (sum codebook^2) * choice[n] * choice[idx[k,n]]
// ---------------------------------------------------------------------------

#define PLANES 128
#define VEC    16
#define KOFF   27

typedef __attribute__((ext_vector_type(16))) __bf16   bf16x16;
typedef __attribute__((ext_vector_type(8)))  float    f32x8;
typedef __attribute__((ext_vector_type(4)))  uint32_t u32x4;
typedef __attribute__((ext_vector_type(8)))  uint32_t u32x8;

union BfTile {
    bf16x16  v;
    uint16_t h[16];
    uint32_t u[8];
    uint4    q[2];
};

__device__ __forceinline__ uint16_t f2bf(float f) {
    unsigned u = __float_as_uint(f);
    u += 0x7FFFu + ((u >> 16) & 1u);      // round-to-nearest-even
    return (uint16_t)(u >> 16);
}
__device__ __forceinline__ float bf2f(uint16_t h) {
    return __uint_as_float(((unsigned)h) << 16);
}

__device__ __forceinline__ f32x8 wmma_bf16(const BfTile& a, const BfTile& b, f32x8 c) {
    // D = A(16x32 bf16) * B(32x16 bf16) + C(16x16 f32)
    return __builtin_amdgcn_wmma_f32_16x16x32_bf16(
        /*neg_a=*/false, a.v, /*neg_b=*/false, b.v,
        /*c_mod=*/(short)0, c, /*reuse_a=*/false, /*reuse_b=*/false);
}

// --- gfx1250 async global->LDS copy (ASYNCcnt path, ISA 08_async_tensor §4)
// Flat LDS addresses carry the wave-relative LDS byte offset in addr[31:0];
// the async op writes LDS at the VDST address plus the instruction offset.
__device__ __forceinline__ uint32_t lds_off_u32(const void* p) {
    return (uint32_t)(uintptr_t)p;
}
__device__ __forceinline__ void async_copy_b128(uint32_t lds_dst, const void* gsrc) {
    asm volatile("global_load_async_to_lds_b128 %0, %1, off"
                 :: "v"(lds_dst), "v"((uint64_t)(uintptr_t)gsrc)
                 : "memory");
}
__device__ __forceinline__ void wait_async0() {
    asm volatile("s_wait_asynccnt 0x0" ::: "memory");
}
__device__ __forceinline__ void wait_tensor0() {
#if __has_builtin(__builtin_amdgcn_s_wait_tensorcnt)
    __builtin_amdgcn_s_wait_tensorcnt(0);
#else
    asm volatile("s_wait_tensorcnt 0x0" ::: "memory");
#endif
}

// --- gfx1250 Tensor Data Mover: DMA a contiguous 32KB (16384 x bf16) weight
// matrix from global into LDS (ISA 08_async_tensor §7/§8, TENSORcnt path).
// 1D descriptor: D# group0 = {count=1, lds_addr, global_addr[56:0], type=2},
// group1 = {data_size=1 (2B), tensor_dim0=16384, tile_dim0=16384}.
// All inputs are workgroup-uniform -> SGPR-resident descriptor groups.
__device__ __forceinline__ void tdm_load_16k_bf16(uint32_t lds_base, const void* gsrc) {
    uint64_t ga = (uint64_t)(uintptr_t)gsrc;
    u32x4 g0;
    g0.x = 1u;                                               // count=1, user D#
    g0.y = lds_base;                                         // lds_addr (bytes)
    g0.z = (uint32_t)ga;                                     // global_addr[31:0]
    g0.w = (uint32_t)((ga >> 32) & 0x01FFFFFFu) | (2u << 30); // addr[56:32] | type=2
    u32x8 g1;
    g1[0] = (1u << 16);          // workgroup_mask=0, data_size=1 (2 bytes)
    g1[1] = (16384u << 16);      // tensor_dim0[15:0] @ bits 63:48
    g1[2] = 0u;                  // tensor_dim0[31:16]=0, tensor_dim1=0
    g1[3] = (16384u << 16);      // tile_dim0 @ bits 127:112
    g1[4] = 0u;                  // tile_dim1=0 (1D), tile_dim2=0
    g1[5] = 0u;                  // tensor_dim0_stride (unused, 1D)
    g1[6] = 0u;
    g1[7] = 0u;
    asm volatile("tensor_load_to_lds %0, %1"
                 :: "s"(g0), "s"(g1)
                 : "memory");
}

// ---------------------------------------------------------------------------
// k_prep: pack weight matrices into WMMA B-operand lane layout (bf16) and
// compute S2 = sum(codebook^2).
//
// B packing used by this code (32x16, contiguous k): lanes 0-15 carry N=lane, K=0..15 (half h -> K=h);
// lanes 16-31 carry N=lane-16, K=16..31. Packed block = 512 halves,
// offset = blk*512 + lane*16 + h  (each lane reads one contiguous 32B chunk).
// ---------------------------------------------------------------------------
__global__ __launch_bounds__(256) void k_prep(
    const float* __restrict__ Wv, const float* __restrict__ Wc,
    const float* __restrict__ Wo, const float* __restrict__ Wq,
    const float* __restrict__ codebook,
    uint16_t* __restrict__ Wv_pk, uint16_t* __restrict__ Wc_pk,
    uint16_t* __restrict__ Wo_pk, uint16_t* __restrict__ Wq_pk,
    float* __restrict__ S2out)
{
    const int TOTSQ = 3 * 128 * 128;          // 3 square matrices
    const int TOTQ  = KOFF * 4 * 512;         // 27 matrices * 4 K-steps * 512
    const int TOT   = TOTSQ + TOTQ;
    int stride = gridDim.x * blockDim.x;
    for (int i = blockIdx.x * blockDim.x + threadIdx.x; i < TOT; i += stride) {
        if (i < TOTSQ) {
            int mat   = i / 16384;
            int local = i - mat * 16384;
            int blk = local >> 9, rem = local & 511;
            int l = rem >> 4, h = rem & 15;
            int kk = blk >> 3, nt = blk & 7;
            int K = kk * 32 + (l >> 4) * 16 + h;
            int c = nt * 16 + (l & 15);
            const float* src = (mat == 0) ? Wv : (mat == 1 ? Wc : Wo);
            uint16_t*    dst = (mat == 0) ? Wv_pk : (mat == 1 ? Wc_pk : Wo_pk);
            dst[local] = f2bf(src[K * PLANES + c]);
        } else {
            int j = i - TOTSQ;
            int blk = j >> 9, rem = j & 511;
            int l = rem >> 4, h = rem & 15;
            int k = blk >> 2, kk = blk & 3;
            int K = kk * 32 + (l >> 4) * 16 + h;
            int c = l & 15;
            Wq_pk[j] = f2bf(Wq[k * (PLANES * VEC) + K * VEC + c]);
        }
    }
    if (blockIdx.x == 0 && threadIdx.x == 0) {
        float s = 0.f;
        #pragma unroll
        for (int c = 0; c < PLANES; ++c) s += codebook[c] * codebook[c];
        S2out[0] = s;
    }
}

// ---------------------------------------------------------------------------
// k_convx: x f32 -> bf16 (packed pairs)
// ---------------------------------------------------------------------------
__global__ __launch_bounds__(256) void k_convx(
    const float* __restrict__ x, uint32_t* __restrict__ xbf, int npairs)
{
    int stride = gridDim.x * blockDim.x;
    for (int i = blockIdx.x * blockDim.x + threadIdx.x; i < npairs; i += stride) {
        float2 f = ((const float2*)x)[i];
        xbf[i] = (uint32_t)f2bf(f.x) | ((uint32_t)f2bf(f.y) << 16);
    }
}

// ---------------------------------------------------------------------------
// k_vgemm: v_f = relu((x@W_v)*g+b) + repeat(coords@W_pos + b_pos, 8)
// One block = 16 rows x 128 cols; wave w owns N-tile w. Store bf16.
//
// B: whole packed 32KB W_v DMA'd into LDS once per workgroup by TDM (wave 0).
// A: 16x128 bf16 tile (4KB) staged via global_load_async_to_lds_b128.
// Both operand streams then read from LDS (ds_load_b128) into the WMMA
// register layouts.
// ---------------------------------------------------------------------------
__global__ __launch_bounds__(256) void k_vgemm(
    const uint16_t* __restrict__ xbf, const uint16_t* __restrict__ Wv_pk,
    const float* __restrict__ vg, const float* __restrict__ vb,
    const float* __restrict__ coords, const float* __restrict__ Wpos,
    const float* __restrict__ bpos, uint16_t* __restrict__ vbf, int N)
{
    __shared__ __align__(16) uint16_t sW[128 * PLANES];  // 32KB packed W_v
    __shared__ __align__(16) uint16_t sA[16 * PLANES];   // 4KB A tile
    __shared__ float sc[16 * 3];
    int t0 = blockIdx.x * 16;
    int tid = threadIdx.x, wv = tid >> 5, l = tid & 31;
    int lm = l & 15, lh = l >> 4;

    if (wv == 0) {                  // one TDM descriptor per workgroup
        tdm_load_16k_bf16(lds_off_u32(sW), Wv_pk);
    }
    // async-stage the A tile: thread t copies bytes [t*16, t*16+16)
    async_copy_b128(lds_off_u32(sA) + tid * 16,
                    xbf + (size_t)t0 * PLANES + tid * 8);
    // prefetch next tile's A chunk into L2 while we compute
    __builtin_prefetch(xbf + (size_t)(t0 + 16) * PLANES + tid * 8, 0, 1);
    if (tid < 48) {
        int rr = t0 + tid / 3;
        sc[tid] = coords[(size_t)min(rr, N - 1) * 3 + (tid % 3)];
    }
    if (wv == 0) wait_tensor0();
    wait_async0();
    __syncthreads();

    const uint16_t* Apt = sA + lm * PLANES + lh * 16;
    f32x8 acc = (f32x8)0.0f;
    #pragma unroll
    for (int kk = 0; kk < 4; ++kk) {
        BfTile a, b;
        a.q[0] = *(const uint4*)(Apt + kk * 32);
        a.q[1] = *(const uint4*)(Apt + kk * 32 + 8);
        const uint16_t* Bp = sW + ((kk * 8 + wv) * 512) + l * 16;
        b.q[0] = ((const uint4*)Bp)[0];
        b.q[1] = ((const uint4*)Bp)[1];
        acc = wmma_bf16(a, b, acc);
    }

    int col = wv * 16 + lm;
    float g = vg[col], be = vb[col];
    int vi = col >> 3;
    float w0 = Wpos[vi], w1 = Wpos[16 + vi], w2 = Wpos[32 + vi], bp = bpos[vi];
    #pragma unroll
    for (int r = 0; r < 8; ++r) {
        int m = lh * 8 + r;                     // C layout: VGPR r -> M = lh*8+r, N = lm
        if (t0 + m < N) {
            float val = acc[r] * g + be;
            val = val > 0.f ? val : 0.f;
            val += sc[m * 3] * w0 + sc[m * 3 + 1] * w1 + sc[m * 3 + 2] * w2 + bp;
            vbf[(size_t)(t0 + m) * PLANES + col] = f2bf(val);
        }
    }
}

// ---------------------------------------------------------------------------
// k_qchoice: sparse 27-offset conv -> q_f (BN+ReLU), then
// choice[n] = sum_j relu((codebook .* repeat(q_f,8)) @ W_choice + b)[n,j]
// Offsets distributed over the 8 waves; cross-wave reduce in LDS.
// Gathered rows are scattered per lane -> direct global b128 register gather.
// ---------------------------------------------------------------------------
__global__ __launch_bounds__(256) void k_qchoice(
    const uint16_t* __restrict__ xbf, const uint16_t* __restrict__ Wq_pk,
    const int* __restrict__ nidx, const int* __restrict__ nmask,
    const float* __restrict__ qg, const float* __restrict__ qb,
    const float* __restrict__ codebook, const uint16_t* __restrict__ Wc_pk,
    const float* __restrict__ bch, float* __restrict__ choice, int N)
{
    __shared__ float red[8][16][16];
    __shared__ float qf[16][16];
    __shared__ float cb[PLANES];
    __shared__ float sums[16];

    int t0 = blockIdx.x * 16;
    int tid = threadIdx.x, wv = tid >> 5, l = tid & 31;
    int lm = l & 15, lh = l >> 4;
    if (tid < PLANES) cb[tid] = codebook[tid];
    if (tid < 16) sums[tid] = 0.f;

    // ---- masked gather-GEMM over this wave's offsets
    int rp = t0 + lm;
    bool inb = rp < N;
    int rpl = inb ? rp : N - 1;
    f32x8 acc = (f32x8)0.0f;
    for (int k = wv; k < KOFF; k += 8) {
        int idx = nidx[(size_t)k * N + rpl];
        int m = inb ? nmask[(size_t)k * N + rpl] : 0;
        const uint16_t* Ap = xbf + (size_t)idx * PLANES + lh * 16;
        #pragma unroll
        for (int kk = 0; kk < 4; ++kk) {
            BfTile a, b;
            if (m) {
                a.q[0] = *(const uint4*)(Ap + kk * 32);
                a.q[1] = *(const uint4*)(Ap + kk * 32 + 8);
            } else {
                a.q[0] = make_uint4(0u, 0u, 0u, 0u);
                a.q[1] = make_uint4(0u, 0u, 0u, 0u);
            }
            const uint16_t* Bp = Wq_pk + (size_t)((k * 4 + kk) * 512) + l * 16;
            b.q[0] = ((const uint4*)Bp)[0];
            b.q[1] = ((const uint4*)Bp)[1];
            acc = wmma_bf16(a, b, acc);
        }
    }
    #pragma unroll
    for (int r = 0; r < 8; ++r) red[wv][lh * 8 + r][lm] = acc[r];
    __syncthreads();

    // ---- reduce 8 wave-partials, BN + ReLU -> q_f tile
    {
        int m = tid >> 4, n = tid & 15;
        float s = 0.f;
        #pragma unroll
        for (int w = 0; w < 8; ++w) s += red[w][m][n];
        s = s * qg[n] + qb[n];
        qf[m][n] = s > 0.f ? s : 0.f;
    }
    __syncthreads();

    // ---- choice GEMM: A[m][c] = cb[c]*qf[m][c/8], wave wv owns N-tile wv
    f32x8 acc2 = (f32x8)0.0f;
    #pragma unroll
    for (int kk = 0; kk < 4; ++kk) {
        BfTile a, b;
        #pragma unroll
        for (int h = 0; h < 16; ++h) {
            int c = kk * 32 + lh * 16 + h;
            a.h[h] = f2bf(cb[c] * qf[lm][c >> 3]);
        }
        const uint16_t* Bp = Wc_pk + (size_t)((kk * 8 + wv) * 512) + l * 16;
        b.q[0] = ((const uint4*)Bp)[0];
        b.q[1] = ((const uint4*)Bp)[1];
        acc2 = wmma_bf16(a, b, acc2);
    }
    int col = wv * 16 + lm;
    float bc = bch[col];
    #pragma unroll
    for (int r = 0; r < 8; ++r) {
        float v = acc2[r] + bc;
        v = v > 0.f ? v : 0.f;
        atomicAdd(&sums[lh * 8 + r], v);        // row-sum across 128 cols
    }
    __syncthreads();
    if (tid < 16 && t0 + tid < N) choice[t0 + tid] = sums[tid];
}

// ---------------------------------------------------------------------------
// k_attn: one wave32 per point. Lane k<27 owns offset k. Rank-1 scores,
// masked softmax via shfl_xor, then weighted gather of v_f (4 ch / lane).
// ---------------------------------------------------------------------------
__global__ __launch_bounds__(256) void k_attn(
    const float* __restrict__ choice, const float* __restrict__ S2p,
    const int* __restrict__ nidx, const int* __restrict__ nmask,
    const uint16_t* __restrict__ vbf, uint16_t* __restrict__ aggbf, int N)
{
    int wv = threadIdx.x >> 5, lane = threadIdx.x & 31;
    int n = blockIdx.x * 8 + wv;
    if (n >= N) return;

    float S2 = S2p[0];
    float cn = choice[n];
    float s = -1e9f;
    int idx = 0, m = 0;
    if (lane < KOFF) {
        idx = nidx[(size_t)lane * N + n];
        m = nmask[(size_t)lane * N + n];
        float ck = choice[idx];
        s = (m != 0) ? (S2 * cn * ck) : -1e9f;
    }
    float mx = s;
    #pragma unroll
    for (int off = 16; off > 0; off >>= 1) mx = fmaxf(mx, __shfl_xor(mx, off, 32));
    float e = (lane < KOFF && m != 0) ? __expf(s - mx) : 0.f;
    float se = e;
    #pragma unroll
    for (int off = 16; off > 0; off >>= 1) se += __shfl_xor(se, off, 32);
    float w = e / se;                           // center offset always valid => se > 0

    float a0 = 0.f, a1 = 0.f, a2 = 0.f, a3 = 0.f;
    #pragma unroll 1
    for (int k = 0; k < KOFF; ++k) {
        float wk = __shfl(w, k, 32);
        int   ik = __shfl(idx, k, 32);
        if (wk > 0.f) {                         // wave-uniform branch
            uint2 d = *(const uint2*)(vbf + (size_t)ik * PLANES + lane * 4);
            a0 += wk * bf2f((uint16_t)(d.x & 0xffffu));
            a1 += wk * bf2f((uint16_t)(d.x >> 16));
            a2 += wk * bf2f((uint16_t)(d.y & 0xffffu));
            a3 += wk * bf2f((uint16_t)(d.y >> 16));
        }
    }
    uint2 o;
    o.x = (uint32_t)f2bf(a0) | ((uint32_t)f2bf(a1) << 16);
    o.y = (uint32_t)f2bf(a2) | ((uint32_t)f2bf(a3) << 16);
    *(uint2*)(aggbf + (size_t)n * PLANES + lane * 4) = o;
}

// ---------------------------------------------------------------------------
// k_outgemm: out = relu((agg@W_out)*g+b) + x   (f32 result)
// B via TDM into LDS, A via async-to-LDS, like k_vgemm.
// ---------------------------------------------------------------------------
__global__ __launch_bounds__(256) void k_outgemm(
    const uint16_t* __restrict__ aggbf, const uint16_t* __restrict__ Wo_pk,
    const float* __restrict__ og, const float* __restrict__ ob,
    const float* __restrict__ x, float* __restrict__ out, int N)
{
    __shared__ __align__(16) uint16_t sW[128 * PLANES];  // 32KB packed W_out
    __shared__ __align__(16) uint16_t sA[16 * PLANES];   // 4KB A tile
    int t0 = blockIdx.x * 16;
    int tid = threadIdx.x, wv = tid >> 5, l = tid & 31;
    int lm = l & 15, lh = l >> 4;

    if (wv == 0) {
        tdm_load_16k_bf16(lds_off_u32(sW), Wo_pk);
    }
    async_copy_b128(lds_off_u32(sA) + tid * 16,
                    aggbf + (size_t)t0 * PLANES + tid * 8);
    __builtin_prefetch(aggbf + (size_t)(t0 + 16) * PLANES + tid * 8, 0, 1);
    if (wv == 0) wait_tensor0();
    wait_async0();
    __syncthreads();

    const uint16_t* Apt = sA + lm * PLANES + lh * 16;
    f32x8 acc = (f32x8)0.0f;
    #pragma unroll
    for (int kk = 0; kk < 4; ++kk) {
        BfTile a, b;
        a.q[0] = *(const uint4*)(Apt + kk * 32);
        a.q[1] = *(const uint4*)(Apt + kk * 32 + 8);
        const uint16_t* Bp = sW + ((kk * 8 + wv) * 512) + l * 16;
        b.q[0] = ((const uint4*)Bp)[0];
        b.q[1] = ((const uint4*)Bp)[1];
        acc = wmma_bf16(a, b, acc);
    }

    int col = wv * 16 + lm;
    float g = og[col], be = ob[col];
    #pragma unroll
    for (int r = 0; r < 8; ++r) {
        int m = lh * 8 + r;
        if (t0 + m < N) {
            float val = acc[r] * g + be;
            val = val > 0.f ? val : 0.f;
            val += x[(size_t)(t0 + m) * PLANES + col];
            out[(size_t)(t0 + m) * PLANES + col] = val;
        }
    }
}

// ---------------------------------------------------------------------------
extern "C" void kernel_launch(void* const* d_in, const int* in_sizes, int n_in,
                              void* d_out, int out_size, void* d_ws, size_t ws_size,
                              hipStream_t stream)
{
    const float* x        = (const float*)d_in[0];
    const float* coords   = (const float*)d_in[1];
    const float* W_q      = (const float*)d_in[2];
    const float* q_gamma  = (const float*)d_in[3];
    const float* q_beta   = (const float*)d_in[4];
    const float* W_v      = (const float*)d_in[5];
    const float* v_gamma  = (const float*)d_in[6];
    const float* v_beta   = (const float*)d_in[7];
    const float* codebook = (const float*)d_in[8];
    const float* W_choice = (const float*)d_in[9];
    const float* b_choice = (const float*)d_in[10];
    const float* W_pos    = (const float*)d_in[11];
    const float* b_pos    = (const float*)d_in[12];
    const float* W_out    = (const float*)d_in[13];
    const float* out_g    = (const float*)d_in[14];
    const float* out_b    = (const float*)d_in[15];
    const int*   nbr_idx  = (const int*)d_in[16];
    const int*   nbr_mask = (const int*)d_in[17];
    float* out = (float*)d_out;

    const int N = in_sizes[0] / PLANES;

    // workspace carve-up (256B aligned)
    uint8_t* ws = (uint8_t*)d_ws;
    size_t off = 0;
    auto carve = [&](size_t bytes) {
        uint8_t* p = ws + off;
        off = (off + bytes + 255) & ~(size_t)255;
        return p;
    };
    uint16_t* xbf    = (uint16_t*)carve((size_t)N * PLANES * 2);
    uint16_t* vbf    = (uint16_t*)carve((size_t)N * PLANES * 2);
    uint16_t* aggbf  = (uint16_t*)carve((size_t)N * PLANES * 2);
    float*    choice = (float*)   carve((size_t)N * 4);
    float*    S2     = (float*)   carve(256);
    uint16_t* Wv_pk  = (uint16_t*)carve(128 * 128 * 2);
    uint16_t* Wc_pk  = (uint16_t*)carve(128 * 128 * 2);
    uint16_t* Wo_pk  = (uint16_t*)carve(128 * 128 * 2);
    uint16_t* Wq_pk  = (uint16_t*)carve(KOFF * 128 * 16 * 2);

    const int tiles = (N + 15) / 16;            // 6250 for N=100000

    k_prep<<<128, 256, 0, stream>>>(W_v, W_choice, W_out, W_q, codebook,
                                    Wv_pk, Wc_pk, Wo_pk, Wq_pk, S2);
    k_convx<<<4096, 256, 0, stream>>>(x, (uint32_t*)xbf, N * PLANES / 2);
    k_vgemm<<<tiles, 256, 0, stream>>>(xbf, Wv_pk, v_gamma, v_beta,
                                       coords, W_pos, b_pos, vbf, N);
    k_qchoice<<<tiles, 256, 0, stream>>>(xbf, Wq_pk, nbr_idx, nbr_mask,
                                         q_gamma, q_beta, codebook, Wc_pk,
                                         b_choice, choice, N);
    k_attn<<<(N + 7) / 8, 256, 0, stream>>>(choice, S2, nbr_idx, nbr_mask,
                                            vbf, aggbf, N);
    k_outgemm<<<tiles, 256, 0, stream>>>(aggbf, Wo_pk, out_g, out_b, x, out, N);
}
